// Attention_18305150616358
// MI455X (gfx1250) — matrix operand order repaired and checked by reading, not compiled, not run on hardware
//
#include <hip/hip_runtime.h>
#include <hip/hip_bf16.h>

typedef __attribute__((ext_vector_type(16))) __bf16 v16bf;
typedef __attribute__((ext_vector_type(8)))  float  v8f;

union ABfrag { v16bf v; unsigned u[8]; };

#define SLEN 2048
#define DMODEL 768
#define NHEAD 12
#define HD 64

// ---------------------------------------------------------------------------
// CDNA5 async global->LDS copy (ASYNCcnt-tracked), per-lane 16B.
// lds_off: raw byte offset into the workgroup LDS segment.
// ---------------------------------------------------------------------------
__device__ __forceinline__ void async_copy_b128(unsigned lds_off, const void* gaddr) {
    asm volatile("global_load_async_to_lds_b128 %0, %1, off"
                 :: "v"(lds_off), "v"((unsigned long long)gaddr)
                 : "memory");
}
__device__ __forceinline__ void wait_async0() {
    asm volatile("s_wait_asynccnt 0" ::: "memory");
}

// ---------------------------------------------------------------------------
// fp32 -> bf16 elementwise convert
// ---------------------------------------------------------------------------
__global__ void f2bf_kernel(const float* __restrict__ in, __bf16* __restrict__ out, size_t n) {
    size_t i = (size_t)blockIdx.x * blockDim.x + threadIdx.x;
    if (i < n) out[i] = (__bf16)in[i];
}

// ---------------------------------------------------------------------------
// Collapse LMF rank-4 factors (sum over rank; "+1" rows fold into biases):
//   wtextb[i][o] = sum_r text_w[r][i][o]                      (bf16 768x768)
//   btext[o]     = sum_r (text_w[r][768][o] + text_b[r][o])
//   latf[b][o]   = sum_i latent[b][i] * sum_r lat_w[r][i][o]
//                + sum_r (lat_w[r][768][o] + lat_b[r][o])
// ---------------------------------------------------------------------------
__global__ void lmf_prep_kernel(const float* __restrict__ tw, const float* __restrict__ tb,
                                const float* __restrict__ lw, const float* __restrict__ lb,
                                const float* __restrict__ latent,
                                __bf16* __restrict__ wtextb, float* __restrict__ btext,
                                float* __restrict__ latf) {
    int i = blockIdx.x * blockDim.x + threadIdx.x;
    if (i < DMODEL * DMODEL) {
        int ii = i / DMODEL, o = i % DMODEL;
        float s = 0.f;
        #pragma unroll
        for (int r = 0; r < 4; ++r) s += tw[((size_t)r * 769 + ii) * DMODEL + o];
        wtextb[(size_t)ii * DMODEL + o] = (__bf16)s;
    }
    if (i < DMODEL) {
        float s = 0.f;
        #pragma unroll
        for (int r = 0; r < 4; ++r)
            s += tw[((size_t)r * 769 + DMODEL) * DMODEL + i] + tb[r * DMODEL + i];
        btext[i] = s;
    }
    if (i < 2 * DMODEL) {
        int b = i / DMODEL, o = i % DMODEL;
        float s = 0.f;
        #pragma unroll
        for (int r = 0; r < 4; ++r)
            s += lw[((size_t)r * 769 + DMODEL) * DMODEL + o] + lb[r * DMODEL + o];
        for (int ii = 0; ii < DMODEL; ++ii) {
            float lv = latent[b * DMODEL + ii];
            float ws = 0.f;
            #pragma unroll
            for (int r = 0; r < 4; ++r) ws += lw[((size_t)r * 769 + ii) * DMODEL + o];
            s += lv * ws;
        }
        latf[i] = s;
    }
}

// ---------------------------------------------------------------------------
// bf16 WMMA GEMM: C[M,N] = A[M,K]*B[K,N] + bias[N]
// 256 threads (8 waves). Block tile 128x128, per-wave 32x64 (2x4 WMMA accs),
// K stepped by 32. Double-buffered LDS; A tile staged with CDNA5 async
// global->LDS B128 copies, B tile transposed into [n][k] via manual scatter.
// MODE 0: f32 out.  MODE 1: QKV split -> bf16 q/k/v.  MODE 2: LMF fuse.
// ---------------------------------------------------------------------------
#define GAS 40                        // padded K-stride (elements) of LDS tiles
#define AS_ELT(buf, r, c)  ((((buf) * 128 + (r)) * GAS) + (c))
#define BS_ELT(buf, n, k)  (2 * 128 * GAS + (((buf) * 128 + (n)) * GAS) + (k))

template <int MODE>
__global__ __launch_bounds__(256)
void gemm_bf16_kernel(const __bf16* __restrict__ A, const __bf16* __restrict__ B,
                      const float* __restrict__ bias, int M, int N, int K,
                      float* __restrict__ outf,
                      __bf16* __restrict__ ob0, __bf16* __restrict__ ob1,
                      __bf16* __restrict__ ob2, const float* __restrict__ latf) {
    __shared__ __align__(16) __bf16 smem[4 * 128 * GAS];   // A[2]+B[2] tiles, 40KB

    const int tid  = threadIdx.x;
    const int lane = tid & 31;
    const int wave = tid >> 5;
    const int m0 = blockIdx.y * 128;
    const int n0 = blockIdx.x * 128;
    const int wm = (wave >> 1) * 32;   // 4 wave-rows
    const int wn = (wave & 1) * 64;    // 2 wave-cols
    const int ml = lane & 15;
    const int hi = lane >> 4;

    v8f acc[2][4] = {};
    const int nk = K / 32;

    // --- stage issue helpers ---
    auto issueA = [&](int buf, int k0) {
        #pragma unroll
        for (int i = 0; i < 2; ++i) {                 // 128x32 bf16 = 512 x 16B
            int idx = tid + i * 256;
            int r = idx >> 2, c16 = idx & 3;
            if (m0 + r < M)
                async_copy_b128((unsigned)(AS_ELT(buf, r, c16 * 8) * 2),
                                A + (size_t)(m0 + r) * K + k0 + c16 * 8);
        }
    };
    auto issueB = [&](int buf, int k0) {
        #pragma unroll
        for (int i = 0; i < 8; ++i) {                 // 32x128 bf16 = 2048 dwords
            int idx = tid + i * 256;
            int k = idx >> 6, n2 = (idx & 63) * 2;
            unsigned val = 0u;
            if (k0 + k < K && n0 + n2 < N)
                val = *(const unsigned*)(B + (size_t)(k0 + k) * N + n0 + n2);
            __bf16 two[2]; *(unsigned*)two = val;
            smem[BS_ELT(buf, n2, k)]     = two[0];
            smem[BS_ELT(buf, n2 + 1, k)] = two[1];
        }
    };

    issueA(0, 0);
    issueB(0, 0);

    for (int s = 0; s < nk; ++s) {
        const int buf = s & 1;
        wait_async0();
        __syncthreads();                              // stage s ready for all waves
        if (s + 1 < nk) {                             // fill other buffer while computing
            issueA(buf ^ 1, (s + 1) * 32);
            issueB(buf ^ 1, (s + 1) * 32);
        }
        if (s + 2 < nk)                               // prefetch B two stages ahead
            __builtin_prefetch(B + (size_t)((s + 2) * 32) * N + n0, 0, 1);

        // --- fragments (documented 16-bit A/B VGPR layouts) ---
        ABfrag af[2], bfr[4];
        #pragma unroll
        for (int t = 0; t < 2; ++t)
            #pragma unroll
            for (int v = 0; v < 8; ++v) {
                int ka = ((v >> 2) << 4) + (hi << 3) + ((v & 3) << 1);
                af[t].u[v] = *(const unsigned*)(&smem[AS_ELT(buf, wm + t * 16 + ml, ka)]);
            }
        #pragma unroll
        for (int t = 0; t < 4; ++t)
            #pragma unroll
            for (int v = 0; v < 8; ++v) {
                int kb = ((v >> 2) << 4) + (hi << 3) + ((v & 3) << 1);
                bfr[t].u[v] = *(const unsigned*)(&smem[BS_ELT(buf, wn + t * 16 + ml, kb)]);
            }
        #pragma unroll
        for (int mi = 0; mi < 2; ++mi)
            #pragma unroll
            for (int ni = 0; ni < 4; ++ni)
                acc[mi][ni] = __builtin_amdgcn_wmma_f32_16x16x32_bf16(
                    false, af[mi].v, false, bfr[ni].v, (short)0, acc[mi][ni], false, false);
        __syncthreads();                              // done reading `buf`
    }

    // --- epilogue; C layout: vgpr r -> m = r + 8*hi, lane -> n = ml ---
    #pragma unroll
    for (int mi = 0; mi < 2; ++mi)
        #pragma unroll
        for (int ni = 0; ni < 4; ++ni)
            #pragma unroll
            for (int r = 0; r < 8; ++r) {
                int row = m0 + wm + mi * 16 + r + hi * 8;
                int col = n0 + wn + ni * 16 + ml;
                if (row < M && col < N) {
                    float v = acc[mi][ni][r] + bias[col];
                    if (MODE == 0) {
                        outf[(size_t)row * N + col] = v;
                    } else if (MODE == 1) {
                        int part = col / DMODEL, c = col % DMODEL;
                        __bf16* dst = (part == 0) ? ob0 : (part == 1) ? ob1 : ob2;
                        dst[(size_t)row * DMODEL + c] = (__bf16)v;
                    } else { // MODE 2: x lat_f[b][col]
                        int b = row >> 11;
                        float f = v * latf[b * DMODEL + col];
                        ob0[(size_t)row * DMODEL + col] = (__bf16)f;
                    }
                }
            }
}

// ---------------------------------------------------------------------------
// Flash attention, causal, per (b, h, 64-row q tile). 4 waves; each wave owns
// 16 q rows, hd=64. K tile staged via async global->LDS; V transposed
// manually; scores and PV via v_wmma_f32_16x16x32_bf16.
// LDS map (elements): Ks[32][72] | Vs[64][36] | Ps[4][16][36]
// ---------------------------------------------------------------------------
#define KS_ELT(j, d)     ((j) * 72 + (d))
#define VS_ELT(d, j)     (32 * 72 + (d) * 36 + (j))
#define PS_ELT(w, m, k)  (32 * 72 + 64 * 36 + ((w) * 16 + (m)) * 36 + (k))

__global__ __launch_bounds__(128)
void attn_kernel(const __bf16* __restrict__ qf, const __bf16* __restrict__ kb,
                 const __bf16* __restrict__ vb, __bf16* __restrict__ ab) {
    __shared__ __align__(16) __bf16 smem[32 * 72 + 64 * 36 + 4 * 16 * 36];

    const int tid = threadIdx.x, lane = tid & 31, wave = tid >> 5;
    const int ml = lane & 15, hi = lane >> 4;
    const int qb0 = blockIdx.x * 64;
    const int h = blockIdx.y;
    const int b = blockIdx.z;
    const int q0w = qb0 + wave * 16;
    const size_t base = (size_t)b * SLEN * DMODEL + (size_t)h * HD;

    // persistent Q A-fragments: 16x64 = two 16x32 frags, straight from global
    ABfrag qa[2];
    #pragma unroll
    for (int c = 0; c < 2; ++c)
        #pragma unroll
        for (int v = 0; v < 8; ++v) {
            int k = ((v >> 2) << 4) + (hi << 3) + ((v & 3) << 1) + c * 32;
            qa[c].u[v] = *(const unsigned*)(qf + base + (size_t)(q0w + ml) * DMODEL + k);
        }

    v8f acc[4] = {};
    float mrow[8], lrow[8];
    #pragma unroll
    for (int r = 0; r < 8; ++r) { mrow[r] = -3.0e38f; lrow[r] = 0.f; }

    const int nb = qb0 / 32 + 2;  // 32-wide key blocks covering j <= qb0+63
    for (int jb = 0; jb < nb; ++jb) {
        const int j0 = jb * 32;
        // --- K tile (32x64) via async copies; V tile transposed manually ---
        #pragma unroll
        for (int i = 0; i < 2; ++i) {                 // 32 rows x 8 x 16B chunks
            int idx = tid + i * 128;
            int j = idx >> 3, c16 = idx & 7;
            async_copy_b128((unsigned)(KS_ELT(j, c16 * 8) * 2),
                            kb + base + (size_t)(j0 + j) * DMODEL + c16 * 8);
        }
        #pragma unroll
        for (int i = 0; i < 8; ++i) {
            int idx = tid + i * 128;
            int j = idx >> 5;
            int d2 = (idx & 31) * 2;
            unsigned vvv = *(const unsigned*)(vb + base + (size_t)(j0 + j) * DMODEL + d2);
            __bf16 two[2]; *(unsigned*)two = vvv;
            smem[VS_ELT(d2, j)]     = two[0];
            smem[VS_ELT(d2 + 1, j)] = two[1];
        }
        wait_async0();
        __syncthreads();

        // --- scores: two 16x16 tiles over j0..j0+31, summing d chunks ---
        v8f s[2] = {};
        #pragma unroll
        for (int t = 0; t < 2; ++t)
            #pragma unroll
            for (int c = 0; c < 2; ++c) {
                ABfrag bf;
                #pragma unroll
                for (int v = 0; v < 8; ++v) {
                    int k = c * 32 + ((v >> 2) << 4) + (hi << 3) + ((v & 3) << 1);
                    bf.u[v] = *(const unsigned*)(&smem[KS_ELT(t * 16 + ml, k)]);
                }
                s[t] = __builtin_amdgcn_wmma_f32_16x16x32_bf16(
                    false, qa[c].v, false, bf.v, (short)0, s[t], false, false);
            }

        // --- scale, causal mask (=-10000 like reference), online softmax ---
        float p0[8], p1[8];
        #pragma unroll
        for (int r = 0; r < 8; ++r) {
            int q = q0w + r + hi * 8;
            float s0 = (j0 + ml      <= q) ? s[0][r] * 0.125f : -10000.f;
            float s1 = (j0 + 16 + ml <= q) ? s[1][r] * 0.125f : -10000.f;
            float mx = fmaxf(s0, s1);
            #pragma unroll
            for (int d = 1; d < 16; d <<= 1) mx = fmaxf(mx, __shfl_xor(mx, d, 32));
            float mnew = fmaxf(mrow[r], mx);
            float scl = __expf(mrow[r] - mnew);
            float e0 = __expf(s0 - mnew);
            float e1 = __expf(s1 - mnew);
            float rs = e0 + e1;
            #pragma unroll
            for (int d = 1; d < 16; d <<= 1) rs += __shfl_xor(rs, d, 32);
            lrow[r] = lrow[r] * scl + rs;
            mrow[r] = mnew;
            #pragma unroll
            for (int t = 0; t < 4; ++t) acc[t][r] = acc[t][r] * scl;
            p0[r] = e0; p1[r] = e1;
        }

        // --- relayout P (C-layout -> A-layout) through per-wave LDS ---
        #pragma unroll
        for (int r = 0; r < 8; ++r) {
            smem[PS_ELT(wave, r + hi * 8, ml)]      = (__bf16)p0[r];
            smem[PS_ELT(wave, r + hi * 8, 16 + ml)] = (__bf16)p1[r];
        }
        ABfrag pa;
        #pragma unroll
        for (int v = 0; v < 8; ++v) {
            int k = ((v >> 2) << 4) + (hi << 3) + ((v & 3) << 1);
            pa.u[v] = *(const unsigned*)(&smem[PS_ELT(wave, ml, k)]);
        }

        // --- acc += P(16x32) x V(32x64), 4 d-tiles of 16 ---
        #pragma unroll
        for (int t = 0; t < 4; ++t) {
            ABfrag bv;
            #pragma unroll
            for (int v = 0; v < 8; ++v) {
                int k = ((v >> 2) << 4) + (hi << 3) + ((v & 3) << 1);
                bv.u[v] = *(const unsigned*)(&smem[VS_ELT(t * 16 + ml, k)]);
            }
            acc[t] = __builtin_amdgcn_wmma_f32_16x16x32_bf16(
                false, pa.v, false, bv.v, (short)0, acc[t], false, false);
        }
        __syncthreads();
    }

    // --- normalize and write merged-head output (bf16) ---
    #pragma unroll
    for (int t = 0; t < 4; ++t)
        #pragma unroll
        for (int r = 0; r < 8; ++r) {
            int q = q0w + r + hi * 8;
            float o = acc[t][r] / lrow[r];
            ab[base + (size_t)q * DMODEL + t * 16 + ml] = (__bf16)o;
        }
}

// ---------------------------------------------------------------------------
// Launch
// ---------------------------------------------------------------------------
extern "C" void kernel_launch(void* const* d_in, const int* in_sizes, int n_in,
                              void* d_out, int out_size, void* d_ws, size_t ws_size,
                              hipStream_t stream) {
    const float* x        = (const float*)d_in[0];   // (2,2048,768)
    const float* latent   = (const float*)d_in[1];   // (2,1,768)
    const float* c_attn_w = (const float*)d_in[2];   // (768,2304)
    const float* c_attn_b = (const float*)d_in[3];   // (2304,)
    const float* c_proj_w = (const float*)d_in[4];   // (768,768)
    const float* c_proj_b = (const float*)d_in[5];   // (768,)
    const float* tw       = (const float*)d_in[6];   // (4,769,768)
    const float* tb       = (const float*)d_in[7];   // (4,768)
    const float* lw       = (const float*)d_in[8];   // (4,769,768)
    const float* lb       = (const float*)d_in[9];   // (4,768)
    float* out = (float*)d_out;

    const int BS = 2 * SLEN;              // 4096 tokens
    size_t off = 0;
    auto carve = [&](size_t bytes) -> void* {
        void* p = (char*)d_ws + off;
        off += (bytes + 255) & ~(size_t)255;
        return p;
    };
    __bf16* xb     = (__bf16*)carve((size_t)BS * DMODEL * 2);
    __bf16* wqkvb  = (__bf16*)carve((size_t)DMODEL * 3 * DMODEL * 2);
    __bf16* wtextb = (__bf16*)carve((size_t)DMODEL * DMODEL * 2);
    __bf16* wprojb = (__bf16*)carve((size_t)DMODEL * DMODEL * 2);
    float*  btext  = (float*)carve(DMODEL * 4);
    float*  latf   = (float*)carve(2 * DMODEL * 4);
    __bf16* qb     = (__bf16*)carve((size_t)BS * DMODEL * 2);
    __bf16* kbuf   = (__bf16*)carve((size_t)BS * DMODEL * 2);
    __bf16* vbuf   = (__bf16*)carve((size_t)BS * DMODEL * 2);
    __bf16* qfb    = (__bf16*)carve((size_t)BS * DMODEL * 2);
    __bf16* ab     = (__bf16*)carve((size_t)BS * DMODEL * 2);

    // 1) downconvert activations + weights to bf16
    {
        size_t n = (size_t)BS * DMODEL;
        f2bf_kernel<<<(unsigned)((n + 255) / 256), 256, 0, stream>>>(x, xb, n);
        n = (size_t)DMODEL * 3 * DMODEL;
        f2bf_kernel<<<(unsigned)((n + 255) / 256), 256, 0, stream>>>(c_attn_w, wqkvb, n);
        n = (size_t)DMODEL * DMODEL;
        f2bf_kernel<<<(unsigned)((n + 255) / 256), 256, 0, stream>>>(c_proj_w, wprojb, n);
    }
    // 2) collapse LMF factors, compute lat_f
    lmf_prep_kernel<<<(DMODEL * DMODEL + 255) / 256, 256, 0, stream>>>(
        tw, tb, lw, lb, latent, wtextb, btext, latf);

    // 3) QKV GEMM: 4096x2304x768, split epilogue -> q/k/v bf16
    gemm_bf16_kernel<1><<<dim3(3 * DMODEL / 128, BS / 128), 256, 0, stream>>>(
        xb, wqkvb, c_attn_b, BS, 3 * DMODEL, DMODEL, nullptr, qb, kbuf, vbuf, nullptr);

    // 4) LMF text GEMM: 4096x768x768, epilogue x lat_f -> fused Q bf16
    gemm_bf16_kernel<2><<<dim3(DMODEL / 128, BS / 128), 256, 0, stream>>>(
        qb, wtextb, btext, BS, DMODEL, DMODEL, nullptr, qfb, nullptr, nullptr, latf);

    // 5) causal flash attention -> merged heads bf16
    attn_kernel<<<dim3(SLEN / 64, NHEAD, 2), 128, 0, stream>>>(qfb, kbuf, vbuf, ab);

    // 6) output projection GEMM: 4096x768x768 -> f32 out
    gemm_bf16_kernel<0><<<dim3(DMODEL / 128, BS / 128), 256, 0, stream>>>(
        ab, wprojb, c_proj_b, BS, DMODEL, DMODEL, out, nullptr, nullptr, nullptr, nullptr);
}
